// MeshConv_79517024518469
// MI455X (gfx1250) — matrix operand order repaired and checked by reading, not compiled, not run on hardware
//
#include <hip/hip_runtime.h>
#include <hip/hip_bf16.h>

typedef __attribute__((ext_vector_type(16))) __bf16 v16bf;
typedef __attribute__((ext_vector_type(8)))  __bf16 v8bf;
typedef __attribute__((ext_vector_type(8)))  float  v8f;

#define WS_COUNTS_OFF   0
#define WS_W1P_OFF      409600      // 256x128 bf16, WMMA-B packed (64 KB)
#define WS_W2P_OFF      475136      // 128x128 bf16, WMMA-B packed (32 KB)

// ---------------------------------------------------------------------------
// Kernel 0: zero out/counts, and pack W1 (first 256 rows) and W2 into the
// exact v_wmma_*_bf16 B-operand lane layout:
//   element j of lane L holds W[k][n] with n = nt*16 + (L&15),
//   k = kb*32 + ((L&16)?8:0) + (j&7) + ((j&8)?16:0)
// ---------------------------------------------------------------------------
__global__ __launch_bounds__(256) void init_pack_kernel(
    const float* __restrict__ W1, const float* __restrict__ W2,
    float* __restrict__ out, float* __restrict__ counts,
    __bf16* __restrict__ W1p, __bf16* __restrict__ W2p,
    int nNodes, long outTotal)
{
  long gid = (long)blockIdx.x * 256 + threadIdx.x;
  if (gid < outTotal) out[gid] = 0.0f;
  if (gid < nNodes)   counts[gid] = 0.0f;
  if (gid < 256 * 128) {
    int k = (int)(gid >> 7), n = (int)(gid & 127);
    int nt = n >> 4, kb = k >> 5, kr = k & 31;
    int j = (kr & 7) | ((kr >> 4) << 3), hib = (kr >> 3) & 1;
    int lane = (n & 15) | (hib << 4);
    W1p[(((nt * 8 + kb) * 32 + lane) << 4) + j] = (__bf16)W1[(k << 7) + n];
  }
  if (gid < 128 * 128) {
    int k = (int)(gid >> 7), n = (int)(gid & 127);
    int nt = n >> 4, kb = k >> 5, kr = k & 31;
    int j = (kr & 7) | ((kr >> 4) << 3), hib = (kr >> 3) & 1;
    int lane = (n & 15) | (hib << 4);
    W2p[(((nt * 4 + kb) * 32 + lane) << 4) + j] = (__bf16)W2[(k << 7) + n];
  }
}

// ---------------------------------------------------------------------------
// Kernel 1: per-edge MLP via WMMA. One wave = 16 edges x 128 outputs.
// ---------------------------------------------------------------------------
__global__ __launch_bounds__(256) void edge_mlp_kernel(
    const float* __restrict__ x, const int* __restrict__ ei,
    const float* __restrict__ ea,
    const float* __restrict__ W1, const float* __restrict__ b1,
    const float* __restrict__ gamma, const float* __restrict__ beta,
    const float* __restrict__ b2,
    const v16bf* __restrict__ W1p, const v16bf* __restrict__ W2p,
    float* __restrict__ out, float* __restrict__ counts,
    int E, int tiles)
{
  __shared__ __align__(16) float  eaLds[8][16][4];
  __shared__ __align__(16) __bf16 actLds[8][16][128];

  const int lane = threadIdx.x & 31;
  const int w    = threadIdx.x >> 5;
  const int tile = blockIdx.x * 8 + w;
  if (tile >= tiles) return;                 // wave-uniform exit; EXEC stays full
  const int e0 = tile * 16;

  const int* __restrict__ srcIdx = ei;       // edge_index[0] = src
  const int* __restrict__ dstIdx = ei + E;   // edge_index[1] = dst

  const int m0  = lane & 15;                 // A row / C column-within-tile
  const int hi  = lane >> 4;
  const int klo = hi * 8;

  const int eM  = (e0 + m0 < E) ? (e0 + m0) : (E - 1);
  const float* __restrict__ rowD = x + (size_t)dstIdx[eM] * 128;
  const float* __restrict__ rowS = x + (size_t)srcIdx[eM] * 128;

  // stage this wave's 16x4 edge_attr tile in LDS
  if (lane < 16) {
    float4 v = {0.f, 0.f, 0.f, 0.f};
    if (e0 + lane < E) v = *(const float4*)&ea[(size_t)(e0 + lane) * 4];
    *(float4*)&eaLds[w][lane][0] = v;
  }
  asm volatile("s_wait_dscnt 0" ::: "memory");

  const v8f zero8 = {0.f, 0.f, 0.f, 0.f, 0.f, 0.f, 0.f, 0.f};
  v8f acc[8];
#pragma unroll
  for (int nt = 0; nt < 8; ++nt) acc[nt] = zero8;

  // ---- GEMM1: [16x256] (x_dst | x_src) @ W1[0:256] ----
#pragma unroll
  for (int half = 0; half < 2; ++half) {
    const float* __restrict__ row = half ? rowS : rowD;
#pragma unroll
    for (int kb = 0; kb < 4; ++kb) {
      const int base = kb * 32 + klo;
      float4 f0 = *(const float4*)(row + base);
      float4 f1 = *(const float4*)(row + base + 4);
      float4 f2 = *(const float4*)(row + base + 16);
      float4 f3 = *(const float4*)(row + base + 20);
      float fv[16] = {f0.x, f0.y, f0.z, f0.w, f1.x, f1.y, f1.z, f1.w,
                      f2.x, f2.y, f2.z, f2.w, f3.x, f3.y, f3.z, f3.w};
      v16bf a;
#pragma unroll
      for (int j = 0; j < 16; ++j) a[j] = (__bf16)fv[j];
      const int kbp = half * 4 + kb;
#pragma unroll
      for (int nt = 0; nt < 8; ++nt) {
        v16bf b = W1p[(nt * 8 + kbp) * 32 + lane];
        acc[nt] = __builtin_amdgcn_wmma_f32_16x16x32_bf16(
            false, a, false, b, (short)0, acc[nt], false, false);
      }
    }
  }

  // ---- edge_attr tail (W1 rows 256..259) + bias b1 ----
#pragma unroll
  for (int nt = 0; nt < 8; ++nt) {
    const int n = nt * 16 + m0;
    const float c0 = W1[256 * 128 + n], c1 = W1[257 * 128 + n];
    const float c2 = W1[258 * 128 + n], c3 = W1[259 * 128 + n];
    const float bb = b1[n];
#pragma unroll
    for (int i = 0; i < 8; ++i) {
      const int m = i + hi * 8;
      float4 e4 = *(const float4*)&eaLds[w][m][0];
      acc[nt][i] += bb + e4.x * c0 + e4.y * c1 + e4.z * c2 + e4.w * c3;
    }
  }

  // ---- GroupNorm (group == 16-wide N tile) + SiLU; write bf16 act to LDS ----
#pragma unroll
  for (int nt = 0; nt < 8; ++nt) {
    const int n = nt * 16 + m0;
    const float g = gamma[n], bet = beta[n];
#pragma unroll
    for (int i = 0; i < 8; ++i) {
      float v = acc[nt][i];
      float s = v, q = v * v;
      s += __shfl_xor(s, 8, 32); q += __shfl_xor(q, 8, 32);
      s += __shfl_xor(s, 4, 32); q += __shfl_xor(q, 4, 32);
      s += __shfl_xor(s, 2, 32); q += __shfl_xor(q, 2, 32);
      s += __shfl_xor(s, 1, 32); q += __shfl_xor(q, 1, 32);
      const float mean = s * 0.0625f;
      const float var  = fmaxf(q * 0.0625f - mean * mean, 0.f);
      float hn = (v - mean) * rsqrtf(var + 1e-5f) * g + bet;
      float act = hn / (1.f + __expf(-hn));
      actLds[w][i + hi * 8][n] = (__bf16)act;
    }
  }
  asm volatile("s_wait_dscnt 0" ::: "memory");

  // ---- GEMM2: act[16x128] @ W2 ----
  v8f acc2[8];
#pragma unroll
  for (int nt = 0; nt < 8; ++nt) acc2[nt] = zero8;

  const v8bf* __restrict__ actRow = (const v8bf*)&actLds[w][m0][0];
#pragma unroll
  for (int kb = 0; kb < 4; ++kb) {
    v8bf lo = actRow[kb * 4 + hi];
    v8bf hh = actRow[kb * 4 + 2 + hi];
    v16bf a = __builtin_shufflevector(lo, hh, 0, 1, 2, 3, 4, 5, 6, 7,
                                      8, 9, 10, 11, 12, 13, 14, 15);
#pragma unroll
    for (int nt = 0; nt < 8; ++nt) {
      v16bf b = W2p[(nt * 4 + kb) * 32 + lane];
      acc2[nt] = __builtin_amdgcn_wmma_f32_16x16x32_bf16(
          false, a, false, b, (short)0, acc2[nt], false, false);
    }
  }

  // ---- bias b2 + atomic scatter into out sums ----
#pragma unroll
  for (int nt = 0; nt < 8; ++nt) {
    const int n = nt * 16 + m0;
    const float bb2 = b2[n];
#pragma unroll
    for (int i = 0; i < 8; ++i) {
      const int m = i + hi * 8;
      const int e = e0 + m;
      if (e < E) {
        const int node = dstIdx[e];
        atomicAdd(&out[(size_t)node * 128 + n], acc2[nt][i] + bb2);
      }
    }
  }
  if (lane < 16 && (e0 + lane) < E)
    atomicAdd(&counts[dstIdx[e0 + lane]], 1.0f);
}

// ---------------------------------------------------------------------------
// Kernel 2: divide sums by max(count, 1)
// ---------------------------------------------------------------------------
__global__ __launch_bounds__(256) void normalize_kernel(
    float* __restrict__ out, const float* __restrict__ counts, long total)
{
  long gid = (long)blockIdx.x * 256 + threadIdx.x;
  if (gid < total) {
    float c = counts[gid >> 7];
    out[gid] = out[gid] / fmaxf(c, 1.0f);
  }
}

extern "C" void kernel_launch(void* const* d_in, const int* in_sizes, int n_in,
                              void* d_out, int out_size, void* d_ws, size_t ws_size,
                              hipStream_t stream) {
  const float* x     = (const float*)d_in[0];
  const int*   ei    = (const int*)d_in[1];
  const float* ea    = (const float*)d_in[2];
  const float* W1    = (const float*)d_in[3];
  const float* b1    = (const float*)d_in[4];
  const float* gamma = (const float*)d_in[5];
  const float* beta  = (const float*)d_in[6];
  const float* W2    = (const float*)d_in[7];
  const float* b2    = (const float*)d_in[8];

  const int nNodes = in_sizes[0] / 128;
  const int E      = in_sizes[1] / 2;
  float* out = (float*)d_out;

  char* wsb = (char*)d_ws;
  float*  counts = (float*)(wsb + WS_COUNTS_OFF);
  __bf16* W1p    = (__bf16*)(wsb + WS_W1P_OFF);
  __bf16* W2p    = (__bf16*)(wsb + WS_W2P_OFF);

  const long outTotal = (long)nNodes * 128;
  const int initBlocks = (int)((outTotal + 255) / 256);

  init_pack_kernel<<<initBlocks, 256, 0, stream>>>(W1, W2, out, counts,
                                                   W1p, W2p, nNodes, outTotal);

  const int tiles  = (E + 15) / 16;
  const int blocks = (tiles + 7) / 8;
  edge_mlp_kernel<<<blocks, 256, 0, stream>>>(
      x, ei, ea, W1, b1, gamma, beta, b2,
      (const v16bf*)(wsb + WS_W1P_OFF), (const v16bf*)(wsb + WS_W2P_OFF),
      out, counts, E, tiles);

  normalize_kernel<<<initBlocks, 256, 0, stream>>>(out, counts, outTotal);
}
